// LSTMGCN_36575941492804
// MI455X (gfx1250) — matrix operand order repaired and checked by reading, not compiled, not run on hardware
//
#include <hip/hip_runtime.h>
#include <hip/hip_bf16.h>
#include <stdint.h>

// ---------------- problem constants (exact tiling) --------------------------
#define NN   20000
#define EE   640000
#define FF   6
#define DD   128
#define GG   512          // 4*D
#define STEPS 11          // 1 + DEPTH
#define LDS_STRIDE 136    // halfwords/row: 272B = 17*16B -> b128-aligned,
                          // bank = 68*r + dw -> conflict-free fragment reads
#define MB   160          // rows per GEMM block (20000 = 125 * 160)
#define MSUB 10           // 160 / 16 m-subtiles
#define SMEM_BYTES (2 * MB * LDS_STRIDE * 2)   // Ahi + Alo = 87040 B (dynamic)

typedef __attribute__((ext_vector_type(16))) __bf16 v16bf;
typedef __attribute__((ext_vector_type(8)))  float  v8f;

// round-to-nearest-even f32 -> bf16 (raw u16)
__device__ __forceinline__ unsigned short bf16_rn(float v) {
    unsigned int u = __float_as_uint(v);
    return (unsigned short)((u + 0x7FFFu + ((u >> 16) & 1u)) >> 16);
}
__device__ __forceinline__ float bf16_to_f32(unsigned short h) {
    return __uint_as_float(((unsigned int)h) << 16);
}
// activation element: u32 = hi_bf16 | (lo_bf16 << 16), value = hi + lo
__device__ __forceinline__ uint32_t pack_split(float v) {
    unsigned short h = bf16_rn(v);
    unsigned short l = bf16_rn(v - bf16_to_f32(h));
    return (uint32_t)h | ((uint32_t)l << 16);
}
__device__ __forceinline__ float unpack_val(uint32_t x) {
    return __uint_as_float(x << 16) + __uint_as_float(x & 0xFFFF0000u);
}

// ---------------------------------------------------------------------------
// Pack one [512 x ldw] fp32 weight matrix (gates = X @ W^T) into B-fragment
// order for v_wmma_f32_16x16x32_bf16, hi/lo interleaved PER LANE so each lane
// reads 64 contiguous bytes (4 x b128) per K-chunk:
//   P[((kc*32 + ntile)*32 + lane)*16 + hl*8 + j]   (u32 = 2 packed bf16)
// B element (k,n): lane = (n&15) + 16*(k>=16), dword j = (k&15)/2.
// ---------------------------------------------------------------------------
__global__ void pack_weights(const float* __restrict__ W, int ldw, int colofs,
                             uint32_t* __restrict__ P) {
    int t = blockIdx.x * blockDim.x + threadIdx.x;     // 0..65535
    int j    = t & 7;
    int hl   = (t >> 3) & 1;
    int lane = (t >> 4) & 31;
    int idx  = t >> 9;             // kc*32 + ntile, 0..127
    int ntg  = idx & 31;
    int kc   = idx >> 5;
    int n    = ntg * 16 + (lane & 15);                 // gate column (0..511)
    int k0   = kc * 32 + ((lane >= 16) ? 16 : 0) + 2 * j;
    float w0 = W[(size_t)n * ldw + colofs + k0];
    float w1 = W[(size_t)n * ldw + colofs + k0 + 1];
    unsigned short h0 = bf16_rn(w0), h1 = bf16_rn(w1);
    unsigned short l0 = bf16_rn(w0 - bf16_to_f32(h0));
    unsigned short l1 = bf16_rn(w1 - bf16_to_f32(h1));
    uint32_t v = hl ? ((uint32_t)l0 | ((uint32_t)l1 << 16))
                    : ((uint32_t)h0 | ((uint32_t)h1 << 16));
    P[((size_t)(idx * 32 + lane) * 16) + hl * 8 + j] = v;
}

// ---------------------------------------------------------------------------
// One input's contribution.  X is PRE-SPLIT packed (hi|lo<<16) per element,
// so staging is a pure deinterleave (v_perm-able), no float math.
// ---------------------------------------------------------------------------
__device__ __forceinline__ void gemm_one(
    const uint32_t* __restrict__ X, const uint32_t* __restrict__ P,
    unsigned short* Ahi, unsigned short* Alo,
    int tid, int lane, int mbase, int ntg, v8f acc[MSUB]) {

    __syncthreads();
    {
        int row = tid >> 4;            // 0..15
        int k0  = (tid & 15) * 8;      // 0..120
#pragma unroll
        for (int s = 0; s < MSUB; ++s) {
            int r = s * 16 + row;
            const uint4* xr = (const uint4*)(X + (size_t)(mbase + r) * DD + k0);
            uint4 a = xr[0], b = xr[1];
            uint32_t xx[8] = {a.x, a.y, a.z, a.w, b.x, b.y, b.z, b.w};
            uint32_t ph[4], pl[4];
#pragma unroll
            for (int j = 0; j < 4; ++j) {
                uint32_t e0 = xx[2 * j], e1 = xx[2 * j + 1];
                ph[j] = (e0 & 0xFFFFu) | (e1 << 16);          // hi halves
                pl[j] = (e0 >> 16) | (e1 & 0xFFFF0000u);      // lo halves
            }
            uint2* dh = (uint2*)&Ahi[r * LDS_STRIDE + k0];    // 8B aligned
            uint2* dl = (uint2*)&Alo[r * LDS_STRIDE + k0];
            dh[0] = make_uint2(ph[0], ph[1]); dh[1] = make_uint2(ph[2], ph[3]);
            dl[0] = make_uint2(pl[0], pl[1]); dl[1] = make_uint2(pl[2], pl[3]);
        }
    }
    __syncthreads();

    const int arow = lane & 15;
    const int koff = (lane >> 4) * 16; // lanes 16-31 hold K 16..31

#pragma unroll
    for (int kc = 0; kc < 4; ++kc) {
        // B fragments: 64 contiguous bytes per lane (hi 32B then lo 32B)
        union { uint4 q[2]; v16bf v; } bh, bl;
        const uint4* pb = (const uint4*)(P + (size_t)((kc * 32 + ntg) * 32 + lane) * 16);
        bh.q[0] = pb[0]; bh.q[1] = pb[1];
        bl.q[0] = pb[2]; bl.q[1] = pb[3];

#pragma unroll
        for (int s = 0; s < MSUB; ++s) {
            union { uint4 q[2]; v16bf v; } ah, al;
            int base = (s * 16 + arow) * LDS_STRIDE + kc * 32 + koff;  // 16B aligned
            const uint4* ph = (const uint4*)&Ahi[base];
            const uint4* pl = (const uint4*)&Alo[base];
            ah.q[0] = ph[0]; ah.q[1] = ph[1];   // +16B = next 8 K
            al.q[0] = pl[0]; al.q[1] = pl[1];
            // Ah*Bh + Ah*Bl + Al*Bh  (~fp32 product precision, f32 accumulate)
            acc[s] = __builtin_amdgcn_wmma_f32_16x16x32_bf16(
                false, ah.v, false, bh.v, (short)0, acc[s], false, false);
            acc[s] = __builtin_amdgcn_wmma_f32_16x16x32_bf16(
                false, ah.v, false, bl.v, (short)0, acc[s], false, false);
            acc[s] = __builtin_amdgcn_wmma_f32_16x16x32_bf16(
                false, al.v, false, bh.v, (short)0, acc[s], false, false);
        }
    }
}

// gates[N,512] = bias + sum_i Xi[N,128] @ Wi^T.  Grid: (125 m-blocks,
// 4 n-blocks), 256 threads; wave w owns n-tile = blockIdx.y*8 + w across 10
// m-subtiles.  Bias (bih+bhh) folded into accumulator init: the whole wave
// tile shares one gate column per lane.
__global__ __launch_bounds__(256, 4) void gemm_gates(
    const uint32_t* __restrict__ X0, const uint32_t* __restrict__ P0,
    const uint32_t* __restrict__ X1, const uint32_t* __restrict__ P1,
    const uint32_t* __restrict__ X2, const uint32_t* __restrict__ P2,
    int ninputs, const float* __restrict__ bsum, float* __restrict__ gates) {

    extern __shared__ unsigned short smem[];
    unsigned short* Ahi = smem;
    unsigned short* Alo = smem + MB * LDS_STRIDE;

    const int tid   = threadIdx.x;
    const int lane  = tid & 31;
    const int wave  = tid >> 5;
    const int mbase = blockIdx.x * MB;
    const int ntg   = blockIdx.y * 8 + wave;   // global n-tile 0..31
    const int col   = ntg * 16 + (lane & 15);  // gate column for this lane

    const float bias = bsum[col];
    v8f acc[MSUB];
#pragma unroll
    for (int s = 0; s < MSUB; ++s)
#pragma unroll
        for (int r = 0; r < 8; ++r) acc[s][r] = bias;

    gemm_one(X0, P0, Ahi, Alo, tid, lane, mbase, ntg, acc);
    if (ninputs > 1) gemm_one(X1, P1, Ahi, Alo, tid, lane, mbase, ntg, acc);
    if (ninputs > 2) gemm_one(X2, P2, Ahi, Alo, tid, lane, mbase, ntg, acc);

    // D layout: vgpr r -> (M = r + 8*(lane>>4), N = lane&15)
#pragma unroll
    for (int s = 0; s < MSUB; ++s) {
        int rb = mbase + s * 16 + ((lane >> 4) << 3);
#pragma unroll
        for (int r = 0; r < 8; ++r)
            gates[(size_t)(rb + r) * GG + col] = acc[s][r];
    }
}

// ------------------------------ helpers ------------------------------------
__global__ void zero_f32(float* __restrict__ p, int count) {
    int t = blockIdx.x * blockDim.x + threadIdx.x;
    if (t < count) p[t] = 0.0f;
}
__global__ void zero_i32(int* __restrict__ p, int count) {
    int t = blockIdx.x * blockDim.x + threadIdx.x;
    if (t < count) p[t] = 0;
}
__global__ void bias_sum(const float* __restrict__ bih, const float* __restrict__ bhh,
                         float* __restrict__ bsum) {
    int t = blockIdx.x * blockDim.x + threadIdx.x;
    if (t < 3 * GG) bsum[t] = bih[t] + bhh[t];
}

// feat[n,d] = relu(features@W_in + b_in), written pre-split packed
__global__ void embed_relu(const float* __restrict__ features,
                           const float* __restrict__ W_in,
                           const float* __restrict__ b_in,
                           uint32_t* __restrict__ feat) {
    int t = blockIdx.x * blockDim.x + threadIdx.x;   // N*D exact
    int d = t & (DD - 1);
    int n = t >> 7;
    float s = b_in[d];
#pragma unroll
    for (int k = 0; k < FF; ++k)
        s += features[(size_t)n * FF + k] * W_in[(size_t)k * DD + d];
    feat[t] = pack_split(fmaxf(s, 0.0f));
}

// ---- CSR build (once per launch): histogram -> scan -> fill ----
__global__ void deg_count(const int* __restrict__ dst, int* __restrict__ cnt) {
    int e = blockIdx.x * blockDim.x + threadIdx.x;
    if (e < EE) atomicAdd(&cnt[dst[e]], 1);
}
__global__ void deg_invert(const int* __restrict__ cnt, float* __restrict__ deginv) {
    int t = blockIdx.x * blockDim.x + threadIdx.x;
    if (t < NN) { int c = cnt[t]; deginv[t] = (c > 0) ? (1.0f / (float)c) : 0.0f; }
}

#define SCAN_T 512
__global__ __launch_bounds__(SCAN_T) void scan_offsets(
    const int* __restrict__ cnt, int* __restrict__ off, int* __restrict__ pos) {
    __shared__ int ssum[SCAN_T];
    const int t  = threadIdx.x;
    const int CH = (NN + SCAN_T - 1) / SCAN_T;   // 40 nodes/thread
    int base = t * CH;
    int s = 0;
    for (int i = 0; i < CH; ++i) {
        int n = base + i;
        if (n < NN) s += cnt[n];
    }
    ssum[t] = s;
    __syncthreads();
    for (int ofs = 1; ofs < SCAN_T; ofs <<= 1) {     // inclusive scan
        int v = (t >= ofs) ? ssum[t - ofs] : 0;
        __syncthreads();
        ssum[t] += v;
        __syncthreads();
    }
    int run = (t == 0) ? 0 : ssum[t - 1];            // exclusive base
    for (int i = 0; i < CH; ++i) {
        int n = base + i;
        if (n < NN) { off[n] = run; pos[n] = run; run += cnt[n]; }
    }
    if (t == SCAN_T - 1) off[NN] = run;
}

__global__ void csr_fill(const int* __restrict__ src, const int* __restrict__ dst,
                         int* __restrict__ pos, int* __restrict__ csr_src) {
    int e = blockIdx.x * blockDim.x + threadIdx.x;
    if (e < EE) {
        int p = atomicAdd(&pos[dst[e]], 1);
        csr_src[p] = src[e];
    }
}

// mean aggregation, one wave per node, no atomics; 1/deg fused.
// x is packed (hi+lo); result re-packed.
__global__ void agg_gather(const uint32_t* __restrict__ x, const int* __restrict__ off,
                           const int* __restrict__ csr_src,
                           const float* __restrict__ deginv,
                           uint32_t* __restrict__ agg) {
    int t = blockIdx.x * blockDim.x + threadIdx.x;
    int n = t >> 5;
    if (n >= NN) return;
    int lane = t & 31;
    int j1 = off[n + 1];
    float s0 = 0.f, s1 = 0.f, s2 = 0.f, s3 = 0.f;
    for (int j = off[n]; j < j1; ++j) {
        const uint4* xr = (const uint4*)(x + (size_t)csr_src[j] * DD);
        uint4 v = xr[lane];
        s0 += unpack_val(v.x); s1 += unpack_val(v.y);
        s2 += unpack_val(v.z); s3 += unpack_val(v.w);
    }
    float di = deginv[n];
    uint4 r = make_uint4(pack_split(s0 * di), pack_split(s1 * di),
                         pack_split(s2 * di), pack_split(s3 * di));
    ((uint4*)(agg + (size_t)n * DD))[lane] = r;
}

__device__ __forceinline__ float sigm(float x) { return 1.0f / (1.0f + __expf(-x)); }

// float4 per thread over N*D/4; h written pre-split packed, c stays fp32.
// Gates already include bias (folded into GEMM accumulator init).
__global__ void lstm_cell(const float* __restrict__ gates,
                          uint32_t* __restrict__ h, float* __restrict__ c) {
    int t = blockIdx.x * blockDim.x + threadIdx.x;   // N*D/4 exact
    int q = t & 31;                                   // float4 index in row
    int n = t >> 5;
    const float4* g = (const float4*)(gates + (size_t)n * GG);
    float4 gi = g[q], gf = g[32 + q], gg = g[64 + q], go = g[96 + q];
    float4 cp = ((const float4*)c)[t];
    float4 cn; uint4 hp;
    {
        float i_ = sigm(gi.x), f_ = sigm(gf.x), o_ = sigm(go.x), g_ = tanhf(gg.x);
        cn.x = f_ * cp.x + i_ * g_; hp.x = pack_split(o_ * tanhf(cn.x));
    }
    {
        float i_ = sigm(gi.y), f_ = sigm(gf.y), o_ = sigm(go.y), g_ = tanhf(gg.y);
        cn.y = f_ * cp.y + i_ * g_; hp.y = pack_split(o_ * tanhf(cn.y));
    }
    {
        float i_ = sigm(gi.z), f_ = sigm(gf.z), o_ = sigm(go.z), g_ = tanhf(gg.z);
        cn.z = f_ * cp.z + i_ * g_; hp.z = pack_split(o_ * tanhf(cn.z));
    }
    {
        float i_ = sigm(gi.w), f_ = sigm(gf.w), o_ = sigm(go.w), g_ = tanhf(gg.w);
        cn.w = f_ * cp.w + i_ * g_; hp.w = pack_split(o_ * tanhf(cn.w));
    }
    ((float4*)c)[t] = cn;
    ((uint4*)h)[t]  = hp;
}

// one wave per node: out[n] = dot(h2[n,:], W_out) + b_out  (h2 packed)
__global__ void out_proj(const uint32_t* __restrict__ h2,
                         const float* __restrict__ W_out,
                         const float* __restrict__ b_out,
                         float* __restrict__ out) {
    int t = blockIdx.x * blockDim.x + threadIdx.x;
    int n = t >> 5;
    if (n >= NN) return;
    int lane = t & 31;
    const uint32_t* hr = h2 + (size_t)n * DD;
    float s = 0.0f;
#pragma unroll
    for (int cI = 0; cI < 4; ++cI)
        s += unpack_val(hr[lane + 32 * cI]) * W_out[lane + 32 * cI];
    for (int off = 16; off > 0; off >>= 1) s += __shfl_down(s, off, 32);
    if (lane == 0) out[n] = s + b_out[0];
}

// ---------------------------------------------------------------------------
extern "C" void kernel_launch(void* const* d_in, const int* in_sizes, int n_in,
                              void* d_out, int out_size, void* d_ws, size_t ws_size,
                              hipStream_t stream) {
    (void)in_sizes; (void)n_in; (void)out_size; (void)ws_size;

    const float* features = (const float*)d_in[0];
    const int*   src      = (const int*)  d_in[1];
    const int*   dst      = (const int*)  d_in[2];
    const float* W_in     = (const float*)d_in[3];
    const float* b_in     = (const float*)d_in[4];
    const float* Wih0     = (const float*)d_in[5];   // [512,256]
    const float* Wih_rest = (const float*)d_in[6];   // [2,512,128]
    const float* Whh      = (const float*)d_in[7];   // [3,512,128]
    const float* bih      = (const float*)d_in[8];   // [3,512]
    const float* bhh      = (const float*)d_in[9];   // [3,512]
    const float* W_out    = (const float*)d_in[10];  // [128]
    const float* b_out    = (const float*)d_in[11];  // [1]
    float* out = (float*)d_out;

    // ---- workspace carve-up (~129 MB); packed u32 == 4B like f32 ----
    const size_t ND = (size_t)NN * DD;
    uint32_t* feat  = (uint32_t*)d_ws;
    uint32_t* agg   = feat + ND;
    uint32_t* hbuf  = agg + ND;          // 3*N*D packed
    float*    cbuf  = (float*)(hbuf + 3 * ND);   // 3*N*D fp32
    float*    gates = cbuf + 3 * ND;     // N*512 fp32
    float*    deginv = gates + (size_t)NN * GG;
    uint32_t* packs = (uint32_t*)(deginv + NN);   // 7 * 65536 u32
    const size_t PSZ = 65536;
    uint32_t* P0 = packs + 0 * PSZ;   // Wih0 cols   0..127 (agg input)
    uint32_t* P1 = packs + 1 * PSZ;   // Wih0 cols 128..255 (feat input)
    uint32_t* P2 = packs + 2 * PSZ;   // Whh[0]
    uint32_t* P3 = packs + 3 * PSZ;   // Wih_rest[0]
    uint32_t* P4 = packs + 4 * PSZ;   // Whh[1]
    uint32_t* P5 = packs + 5 * PSZ;   // Wih_rest[1]
    uint32_t* P6 = packs + 6 * PSZ;   // Whh[2]
    int* cnt     = (int*)(packs + 7 * PSZ);
    int* pos     = cnt + NN;
    int* off     = pos + NN;             // NN+1
    int* csr_src = off + NN + 1;         // EE
    float* bsum  = (float*)(csr_src + EE);        // 3*512

    const int T = 256;
    const int ND_BLK  = (int)(ND / T);         // 10000
    const int ND4_BLK = (int)(ND / 4 / T);     // 2500
    const int HC_BLK  = (int)(6 * ND / T);     // 60000

    // ---- one-time prep ----
    zero_f32<<<HC_BLK, T, 0, stream>>>((float*)hbuf, (int)(6 * ND)); // h(pack=0.0),c
    zero_i32<<<(NN + T - 1) / T, T, 0, stream>>>(cnt, NN);
    deg_count<<<EE / T, T, 0, stream>>>(dst, cnt);
    deg_invert<<<(NN + T - 1) / T, T, 0, stream>>>(cnt, deginv);
    scan_offsets<<<1, SCAN_T, 0, stream>>>(cnt, off, pos);
    csr_fill<<<EE / T, T, 0, stream>>>(src, dst, pos, csr_src);
    bias_sum<<<(3 * GG) / T, T, 0, stream>>>(bih, bhh, bsum);

    pack_weights<<<256, T, 0, stream>>>(Wih0, 256, 0,   P0);
    pack_weights<<<256, T, 0, stream>>>(Wih0, 256, 128, P1);
    pack_weights<<<256, T, 0, stream>>>(Whh + 0 * GG * DD, 128, 0, P2);
    pack_weights<<<256, T, 0, stream>>>(Wih_rest + 0 * GG * DD, 128, 0, P3);
    pack_weights<<<256, T, 0, stream>>>(Whh + 1 * GG * DD, 128, 0, P4);
    pack_weights<<<256, T, 0, stream>>>(Wih_rest + 1 * GG * DD, 128, 0, P5);
    pack_weights<<<256, T, 0, stream>>>(Whh + 2 * GG * DD, 128, 0, P6);

    embed_relu<<<ND_BLK, T, 0, stream>>>(features, W_in, b_in, feat);

    uint32_t* h0 = hbuf;            float* c0 = cbuf;
    uint32_t* h1 = hbuf + ND;       float* c1 = cbuf + ND;
    uint32_t* h2 = hbuf + 2 * ND;   float* c2 = cbuf + 2 * ND;

    const dim3 ggrid(125, 4);    // 125 m-blocks x 4 n-blocks

    for (int step = 0; step < STEPS; ++step) {
        const uint32_t* xin = (step == 0) ? feat : h2;   // carry = prev h2

        // mean aggregation over incoming edges (CSR gather, no atomics)
        agg_gather<<<NN * 32 / T, T, 0, stream>>>(xin, off, csr_src, deginv, agg);

        // layer 0: gates = bias0 + agg@W0a^T + xin@W0b^T + h0@Whh0^T
        gemm_gates<<<ggrid, T, SMEM_BYTES, stream>>>(agg, P0, xin, P1, h0, P2, 3,
                                                     bsum + 0 * GG, gates);
        lstm_cell<<<ND4_BLK, T, 0, stream>>>(gates, h0, c0);

        // layer 1
        gemm_gates<<<ggrid, T, SMEM_BYTES, stream>>>(h0, P3, h1, P4, nullptr, nullptr, 2,
                                                     bsum + 1 * GG, gates);
        lstm_cell<<<ND4_BLK, T, 0, stream>>>(gates, h1, c1);

        // layer 2
        gemm_gates<<<ggrid, T, SMEM_BYTES, stream>>>(h1, P5, h2, P6, nullptr, nullptr, 2,
                                                     bsum + 2 * GG, gates);
        lstm_cell<<<ND4_BLK, T, 0, stream>>>(gates, h2, c2);
    }

    out_proj<<<NN * 32 / T, T, 0, stream>>>(h2, W_out, b_out, out);
}
